// BitNetV3TransformerBlock_10419590660567
// MI455X (gfx1250) — matrix operand order repaired and checked by reading, not compiled, not run on hardware
//
#include <hip/hip_runtime.h>

typedef unsigned short u16;
typedef __attribute__((ext_vector_type(16))) __bf16 v16bf;
typedef __attribute__((ext_vector_type(8)))  float  v8f;
typedef __attribute__((ext_vector_type(4)))  unsigned int v4u;

#define B_DIM  2
#define S_LEN  2048
#define D_DIM  2048
#define H_NUM  16
#define HD_DIM 128
#define DFF    8192
#define M_TOK  (B_DIM * S_LEN)   // 4096

static __device__ __forceinline__ u16 f2bf(float f) {
    unsigned u = __float_as_uint(f);
    return (u16)((u + 0x7FFFu + ((u >> 16) & 1u)) >> 16);
}

static __device__ __forceinline__ v8f zero8() {
    v8f z;
#pragma unroll
    for (int i = 0; i < 8; ++i) z[i] = 0.0f;
    return z;
}

// Load a 32-byte contiguous B fragment (caller applies the lane-half K offset).
static __device__ __forceinline__ v16bf ld16(const u16* p) {
    union { v4u q[2]; v16bf v; } r;
    r.q[0] = *(const v4u*)(p);
    r.q[1] = *(const v4u*)(p + 8);
    return r.v;
}

// A-fragment, contiguous K order: lane half selects K base 0/16,
// two 16B chunks at +0..7 and +8..15 relative to that base.
static __device__ __forceinline__ v16bf ld_a(const u16* rowbase, unsigned ab) {
    union { v4u q[2]; v16bf v; } r;
    r.q[0] = *(const v4u*)(rowbase + ab);
    r.q[1] = *(const v4u*)(rowbase + ab + 8);
    return r.v;
}

static __device__ __forceinline__ v8f wmma_bf16(v16bf a, v16bf b, v8f c) {
    return __builtin_amdgcn_wmma_f32_16x16x32_bf16(
        false, a, false, b, (short)0, c, false, false);
}

// Async copy 16B/lane from global to LDS (ASYNCcnt-tracked DMA path).
// lds = wave-relative LDS byte address, g = per-lane global address.
static __device__ __forceinline__ void async_cp16(unsigned lds, const void* g) {
    asm volatile("global_load_async_to_lds_b128 %0, %1, off"
                 :: "v"(lds), "v"(g) : "memory");
}
static __device__ __forceinline__ void wait_async0() {
    asm volatile("s_wait_asynccnt 0x0" ::: "memory");
}

// ---------------------------------------------------------------- elementwise

__global__ void cvt_kernel(const float* __restrict__ in, u16* __restrict__ out,
                           size_t n) {
    size_t i = (size_t)blockIdx.x * blockDim.x + threadIdx.x;
    size_t stride = (size_t)gridDim.x * blockDim.x;
    for (; i < n; i += stride) out[i] = f2bf(in[i]);
}

__global__ void silu_mul_kernel(const float* __restrict__ g,
                                const float* __restrict__ u,
                                u16* __restrict__ out, size_t n) {
    size_t i = (size_t)blockIdx.x * blockDim.x + threadIdx.x;
    size_t stride = (size_t)gridDim.x * blockDim.x;
    for (; i < n; i += stride) {
        float gv = g[i];
        float s = gv / (1.0f + __expf(-gv));
        out[i] = f2bf(s * u[i]);
    }
}

__global__ __launch_bounds__(256) void rmsnorm_kernel(
    const float* __restrict__ x, const float* __restrict__ w,
    u16* __restrict__ out, int Dd) {
    int row = blockIdx.x;
    const float* xr = x + (size_t)row * Dd;
    float s = 0.0f;
    for (int i = threadIdx.x; i < Dd; i += 256) {
        float v = xr[i];
        s += v * v;
    }
    __shared__ float red[256];
    red[threadIdx.x] = s;
    __syncthreads();
    for (int k = 128; k; k >>= 1) {
        if ((int)threadIdx.x < k) red[threadIdx.x] += red[threadIdx.x + k];
        __syncthreads();
    }
    float rstd = rsqrtf(red[0] / (float)Dd + 1e-6f);
    u16* orow = out + (size_t)row * Dd;
    for (int i = threadIdx.x; i < Dd; i += 256) orow[i] = f2bf(xr[i] * rstd * w[i]);
}

// -------------------------------------------------------------------- GEMM
// C[M,N] = A[M,K](bf16) @ W[N,K](bf16)^T (+ add), f32 out.
// 8 waves/block, wave tile 64x64, block tile 128x256, K step 32.
// Operands staged through double-buffered LDS via async global->LDS DMA;
// WMMA fragments read from LDS (padded stride 40 u16 = conflict-free).

#define LSTR 40  // padded row stride (u16 units)

__global__ __launch_bounds__(256) void gemm_bf16_kernel(
    const u16* __restrict__ A, const u16* __restrict__ W,
    const float* __restrict__ add, float* __restrict__ C,
    int Mm, int Nn, int Kk) {
    __shared__ __align__(16) u16 AsBuf[2][128 * LSTR];
    __shared__ __align__(16) u16 WsBuf[2][256 * LSTR];

    unsigned tid = threadIdx.x;
    unsigned wave = tid >> 5, lane = tid & 31;
    unsigned wm = wave >> 2, wn = wave & 3;
    unsigned hlf = (lane >> 4) & 1;
    unsigned ab = hlf ? 16u : 0u;   // A-fragment K base
    unsigned bb = hlf ? 16u : 0u;   // B-fragment K base
    int m0b = blockIdx.y * 128, n0b = blockIdx.x * 256;
    int m0w = (int)wm * 64, n0w = (int)wn * 64;

    // Per-thread async staging descriptors: 16B chunks.
    // A tile: 128 rows x 32 K = 512 chunks; W tile: 256 rows x 32 K = 1024.
    unsigned aLds[2]; const u16* aG[2];
#pragma unroll
    for (int j = 0; j < 2; ++j) {
        unsigned c = tid + (unsigned)j * 256u;
        unsigned row = c >> 2, part = c & 3;
        aLds[j] = (unsigned)(size_t)&AsBuf[0][row * LSTR + part * 8];
        aG[j] = A + (size_t)(m0b + (int)row) * Kk + part * 8;
    }
    unsigned wLds[4]; const u16* wG[4];
#pragma unroll
    for (int j = 0; j < 4; ++j) {
        unsigned c = tid + (unsigned)j * 256u;
        unsigned row = c >> 2, part = c & 3;
        wLds[j] = (unsigned)(size_t)&WsBuf[0][row * LSTR + part * 8];
        wG[j] = W + (size_t)(n0b + (int)row) * Kk + part * 8;
    }
    const unsigned AS_B = (unsigned)sizeof(AsBuf[0]);
    const unsigned WS_B = (unsigned)sizeof(WsBuf[0]);

    v8f acc[4][4];
#pragma unroll
    for (int mi = 0; mi < 4; ++mi)
#pragma unroll
        for (int ni = 0; ni < 4; ++ni) acc[mi][ni] = zero8();

    // Prologue: stage k=0 into buffer 0.
#pragma unroll
    for (int j = 0; j < 2; ++j) async_cp16(aLds[j], aG[j]);
#pragma unroll
    for (int j = 0; j < 4; ++j) async_cp16(wLds[j], wG[j]);

    unsigned buf = 0;
    for (int k = 0; k < Kk; k += 32, buf ^= 1) {
        wait_async0();      // my copies into buf have landed
        __syncthreads();    // everyone's copies landed; prev-buf readers done

        if (k + 32 < Kk) {  // prefetch next K-slice into the other buffer
            unsigned nb = buf ^ 1;
            int k2 = k + 32;
#pragma unroll
            for (int j = 0; j < 2; ++j)
                async_cp16(aLds[j] + nb * AS_B, aG[j] + k2);
#pragma unroll
            for (int j = 0; j < 4; ++j)
                async_cp16(wLds[j] + nb * WS_B, wG[j] + k2);
        }

        const u16* as = &AsBuf[buf][0];
        const u16* ws = &WsBuf[buf][0];
        v16bf af[4], bfm[4];
#pragma unroll
        for (int mi = 0; mi < 4; ++mi)
            af[mi] = ld_a(as + (m0w + mi * 16 + (int)(lane & 15)) * LSTR, ab);
#pragma unroll
        for (int ni = 0; ni < 4; ++ni)
            bfm[ni] = ld16(ws + (n0w + ni * 16 + (int)(lane & 15)) * LSTR + bb);
#pragma unroll
        for (int mi = 0; mi < 4; ++mi)
#pragma unroll
            for (int ni = 0; ni < 4; ++ni)
                acc[mi][ni] = wmma_bf16(af[mi], bfm[ni], acc[mi][ni]);
    }

#pragma unroll
    for (int mi = 0; mi < 4; ++mi)
#pragma unroll
        for (int ni = 0; ni < 4; ++ni)
#pragma unroll
            for (int r = 0; r < 8; ++r) {
                int m = m0b + m0w + mi * 16 + (int)hlf * 8 + r;
                int n = n0b + n0w + ni * 16 + (int)(lane & 15);
                size_t idx = (size_t)m * Nn + n;
                float v = acc[mi][ni][r];
                if (add) v += add[idx];
                C[idx] = v;
            }
}

// --------------------------------------------------------------- attention
// Flash-style causal attention. Block = 128 threads (4 waves); block handles
// 64 query rows of one (b,h); each wave owns 16 query rows. Keys processed in
// chunks of 32 with online softmax. Q/K/V are bf16 views of [B,S,D] with the
// head slice at column h*HD.

#define VT_STRIDE 40  // padded key-stride for V^T in LDS (conflict-free, 16B)
#define P_STRIDE  40

__global__ __launch_bounds__(128) void attn_kernel(
    const u16* __restrict__ qb, const u16* __restrict__ kb,
    const u16* __restrict__ vb, float* __restrict__ o) {
    __shared__ __align__(16) u16 VT[HD_DIM * VT_STRIDE];
    __shared__ __align__(16) u16 Pb[4][16 * P_STRIDE];

    unsigned tid = threadIdx.x;
    unsigned wave = tid >> 5, lane = tid & 31;
    unsigned hlf = (lane >> 4) & 1;
    unsigned ab = hlf ? 16u : 0u;
    unsigned bb = hlf ? 16u : 0u;

    int bh = blockIdx.y;
    int b = bh / H_NUM, h = bh % H_NUM;
    int qblk = blockIdx.x * 64;
    int qBase = qblk + (int)wave * 16;
    const size_t headOff = ((size_t)b * S_LEN) * D_DIM + (size_t)h * HD_DIM;
    const float SCALE = 0.08838834764831845f;  // 1/sqrt(128)
    const float NEG = -1e30f;

    // Q A-fragments for all 4 K-chunks of HD (kept in registers).
    v16bf aq[4];
    {
        const u16* qrow = qb + headOff + (size_t)(qBase + (int)(lane & 15)) * D_DIM;
#pragma unroll
        for (int c = 0; c < 4; ++c) aq[c] = ld_a(qrow + c * 32, ab);
    }

    v8f oacc[8];
#pragma unroll
    for (int nt = 0; nt < 8; ++nt) oacc[nt] = zero8();
    float m_r[8], l_r[8];
#pragma unroll
    for (int r = 0; r < 8; ++r) { m_r[r] = NEG; l_r[r] = 0.0f; }

    int kEnd = qblk + 64;
    for (int k0 = 0; k0 < kEnd; k0 += 32) {
        // ---- stage V^T chunk (32 keys x 128 hd) into LDS
        __syncthreads();  // protect previous iteration's V^T reads
        for (int e = tid; e < 32 * HD_DIM; e += 128) {
            int key = e >> 7;
            int hd = e & 127;
            VT[hd * VT_STRIDE + key] =
                vb[headOff + (size_t)(k0 + key) * D_DIM + hd];
        }
        __syncthreads();

        // ---- scores: 16x32 = Q(16x128) @ K(32x128)^T
        v8f st[2];
        st[0] = zero8();
        st[1] = zero8();
#pragma unroll
        for (int ti = 0; ti < 2; ++ti) {
            int keyc = k0 + ti * 16 + (int)(lane & 15);
            const u16* krow = kb + headOff + (size_t)keyc * D_DIM + bb;
#pragma unroll
            for (int c = 0; c < 4; ++c) {
                v16bf bk = ld16(krow + c * 32);
                st[ti] = wmma_bf16(aq[c], bk, st[ti]);
            }
        }

        // ---- scale + causal mask + online softmax
        float mx[8];
#pragma unroll
        for (int r = 0; r < 8; ++r) {
            int qrow = qBase + (int)hlf * 8 + r;
            float s0 = st[0][r] * SCALE;
            float s1 = st[1][r] * SCALE;
            if (k0 + (int)(lane & 15) > qrow) s0 = NEG;
            if (k0 + 16 + (int)(lane & 15) > qrow) s1 = NEG;
            st[0][r] = s0;
            st[1][r] = s1;
            float m = fmaxf(s0, s1);
#pragma unroll
            for (int off = 8; off; off >>= 1)
                m = fmaxf(m, __shfl_xor(m, off, 32));
            mx[r] = m;
        }
        float scl[8];
#pragma unroll
        for (int r = 0; r < 8; ++r) {
            float mnew = fmaxf(m_r[r], mx[r]);
            float sc = __expf(m_r[r] - mnew);
            m_r[r] = mnew;
            float p0 = __expf(st[0][r] - mnew);
            float p1 = __expf(st[1][r] - mnew);
            st[0][r] = p0;
            st[1][r] = p1;
            float srow = p0 + p1;
#pragma unroll
            for (int off = 8; off; off >>= 1)
                srow += __shfl_xor(srow, off, 32);
            l_r[r] = l_r[r] * sc + srow;
            scl[r] = sc;
        }
#pragma unroll
        for (int nt = 0; nt < 8; ++nt)
#pragma unroll
            for (int r = 0; r < 8; ++r) oacc[nt][r] *= scl[r];

        // ---- write P (C-layout -> row-major bf16 in LDS)
        u16* pw = &Pb[wave][0];
#pragma unroll
        for (int ti = 0; ti < 2; ++ti)
#pragma unroll
            for (int r = 0; r < 8; ++r)
                pw[((int)hlf * 8 + r) * P_STRIDE + ti * 16 + (int)(lane & 15)] =
                    f2bf(st[ti][r]);
        asm volatile("s_wait_dscnt 0" ::: "memory");

        // ---- P(16x32) @ V(32x128): read P as A-fragment, V^T as B-fragments
        v16bf pa = ld_a(pw + (int)(lane & 15) * P_STRIDE, ab);
#pragma unroll
        for (int nt = 0; nt < 8; ++nt) {
            v16bf bv = ld16(&VT[(nt * 16 + (int)(lane & 15)) * VT_STRIDE + bb]);
            oacc[nt] = wmma_bf16(pa, bv, oacc[nt]);
        }
    }

    // ---- epilogue: normalize and store f32 [B,S,D]
    float inv[8];
#pragma unroll
    for (int r = 0; r < 8; ++r) inv[r] = 1.0f / l_r[r];
#pragma unroll
    for (int nt = 0; nt < 8; ++nt)
#pragma unroll
        for (int r = 0; r < 8; ++r) {
            int row = qBase + (int)hlf * 8 + r;
            int col = h * HD_DIM + nt * 16 + (int)(lane & 15);
            o[((size_t)b * S_LEN + row) * D_DIM + col] = oacc[nt][r] * inv[r];
        }
}

// ----------------------------------------------------------------- launcher

#define MiB(x) ((size_t)(x) * 1024 * 1024)

extern "C" void kernel_launch(void* const* d_in, const int* in_sizes, int n_in,
                              void* d_out, int out_size, void* d_ws,
                              size_t ws_size, hipStream_t stream) {
    (void)in_sizes; (void)n_in; (void)out_size; (void)ws_size;
    const float* hs  = (const float*)d_in[0];
    const float* wq  = (const float*)d_in[1];
    const float* wk  = (const float*)d_in[2];
    const float* wv  = (const float*)d_in[3];
    const float* wo  = (const float*)d_in[4];
    const float* wg  = (const float*)d_in[5];
    const float* wu  = (const float*)d_in[6];
    const float* wd  = (const float*)d_in[7];
    const float* ln1 = (const float*)d_in[8];
    const float* ln2 = (const float*)d_in[9];
    float* out = (float*)d_out;

    char* w = (char*)d_ws;
    // persistent
    u16*   wq_b  = (u16*)(w + MiB(0));
    u16*   wk_b  = (u16*)(w + MiB(8));
    u16*   wv_b  = (u16*)(w + MiB(16));
    u16*   wo_b  = (u16*)(w + MiB(24));
    u16*   wg_b  = (u16*)(w + MiB(32));
    u16*   wu_b  = (u16*)(w + MiB(64));
    u16*   wd_b  = (u16*)(w + MiB(96));
    float* hid_f = (float*)(w + MiB(128));
    u16*   y_b   = (u16*)(w + MiB(160));
    u16*   h_b   = (u16*)(w + MiB(176));
    // temporaries (dead after the attention/wo stage)
    u16*   x_b   = (u16*)(w + MiB(240));
    float* q_f   = (float*)(w + MiB(256));
    float* k_f   = (float*)(w + MiB(288));
    float* v_f   = (float*)(w + MiB(320));
    u16*   q_b   = (u16*)(w + MiB(352));
    u16*   k_b   = (u16*)(w + MiB(368));
    u16*   v_b   = (u16*)(w + MiB(384));
    float* ao_f  = (float*)(w + MiB(400));
    u16*   ao_b  = (u16*)(w + MiB(432));
    // MLP reuses the dead temp region
    float* gate_f = (float*)(w + MiB(240));
    float* up_f   = (float*)(w + MiB(368));

    const size_t nDD = (size_t)D_DIM * D_DIM;
    const size_t nFD = (size_t)DFF * D_DIM;
    const size_t nMD = (size_t)M_TOK * D_DIM;
    const size_t nMF = (size_t)M_TOK * DFF;

    dim3 blk256(256), blk128(128);
    dim3 cvg(4096);

    // 1) weights -> bf16
    cvt_kernel<<<cvg, blk256, 0, stream>>>(wq, wq_b, nDD);
    cvt_kernel<<<cvg, blk256, 0, stream>>>(wk, wk_b, nDD);
    cvt_kernel<<<cvg, blk256, 0, stream>>>(wv, wv_b, nDD);
    cvt_kernel<<<cvg, blk256, 0, stream>>>(wo, wo_b, nDD);
    cvt_kernel<<<cvg, blk256, 0, stream>>>(wg, wg_b, nFD);
    cvt_kernel<<<cvg, blk256, 0, stream>>>(wu, wu_b, nFD);
    cvt_kernel<<<cvg, blk256, 0, stream>>>(wd, wd_b, nFD);

    // 2) x = rmsnorm(hidden, ln1) -> bf16
    rmsnorm_kernel<<<dim3(M_TOK), blk256, 0, stream>>>(hs, ln1, x_b, D_DIM);

    // 3) q/k/v projections
    dim3 gD(D_DIM / 256, M_TOK / 128);
    gemm_bf16_kernel<<<gD, blk256, 0, stream>>>(x_b, wq_b, nullptr, q_f, M_TOK, D_DIM, D_DIM);
    gemm_bf16_kernel<<<gD, blk256, 0, stream>>>(x_b, wk_b, nullptr, k_f, M_TOK, D_DIM, D_DIM);
    gemm_bf16_kernel<<<gD, blk256, 0, stream>>>(x_b, wv_b, nullptr, v_f, M_TOK, D_DIM, D_DIM);
    cvt_kernel<<<cvg, blk256, 0, stream>>>(q_f, q_b, nMD);
    cvt_kernel<<<cvg, blk256, 0, stream>>>(k_f, k_b, nMD);
    cvt_kernel<<<cvg, blk256, 0, stream>>>(v_f, v_b, nMD);

    // 4) causal flash attention
    attn_kernel<<<dim3(S_LEN / 64, B_DIM * H_NUM), blk128, 0, stream>>>(q_b, k_b, v_b, ao_f);
    cvt_kernel<<<cvg, blk256, 0, stream>>>(ao_f, ao_b, nMD);

    // 5) out proj + residual
    gemm_bf16_kernel<<<gD, blk256, 0, stream>>>(ao_b, wo_b, hs, hid_f, M_TOK, D_DIM, D_DIM);

    // 6) y = rmsnorm(hidden, ln2) -> bf16
    rmsnorm_kernel<<<dim3(M_TOK), blk256, 0, stream>>>(hid_f, ln2, y_b, D_DIM);

    // 7) MLP
    dim3 gF(DFF / 256, M_TOK / 128);
    gemm_bf16_kernel<<<gF, blk256, 0, stream>>>(y_b, wg_b, nullptr, gate_f, M_TOK, DFF, D_DIM);
    gemm_bf16_kernel<<<gF, blk256, 0, stream>>>(y_b, wu_b, nullptr, up_f, M_TOK, DFF, D_DIM);
    silu_mul_kernel<<<cvg, blk256, 0, stream>>>(gate_f, up_f, h_b, nMF);
    gemm_bf16_kernel<<<gD, blk256, 0, stream>>>(h_b, wd_b, hid_f, out, M_TOK, D_DIM, DFF);
}
